// MultiHeadSelfAttention_6193342841061
// MI455X (gfx1250) — matrix operand order repaired and checked by reading, not compiled, not run on hardware
//
#include <hip/hip_runtime.h>
#include <hip/hip_bf16.h>

// ---------------------------------------------------------------------------
// MultiHeadSelfAttention for MI455X (gfx1250, wave32, WMMA)
//   D=256, H=8, B=4, S=2048.  All GEMMs run through v_wmma_f32_16x16x32_bf16
//   (bf16 inputs, f32 accumulate). Softmax is over the QUERY axis (reference
//   quirk) -> column-wise normalization of each SxS score matrix.
//   Each wave owns a 32x64 output block: 2x4 WMMA tiles, 8 f32 accumulators,
//   12 b128 loads : 8 WMMAs per k-step.
// ---------------------------------------------------------------------------

typedef __attribute__((ext_vector_type(16))) __bf16 v16bf;
typedef __attribute__((ext_vector_type(8)))  __bf16 v8bf;
typedef __attribute__((ext_vector_type(8)))  float  v8f;

namespace {
constexpr int gB = 4;
constexpr int gS = 2048;
constexpr int gD = 256;
constexpr int gH = 8;
constexpr int gHD = gH * gD;  // 2048
}

#define DEVFN static __device__ __forceinline__

// --- WMMA fragment loaders (layouts per CDNA5 ISA §7.12.2) ------
// A matrix 16x32 bf16: row = lane&15; lanes 0-15 hold K 0..7 & 16..23,
// lanes 16-31 hold K 8..15 & 24..31.  `p` is the per-lane resolved pointer:
//   p = tile_origin + (lane&15)*ld + ((lane>>4)<<3)
DEVFN v16bf frag_a(const __bf16* p) {
  v8bf lo = *(const v8bf*)(p);        // K = kb .. kb+7
  v8bf hi = *(const v8bf*)(p + 16);   // K = kb+16 .. kb+23
  v16bf r;
#pragma unroll
  for (int i = 0; i < 8; ++i) { r[i] = lo[i]; r[i + 8] = hi[i]; }
  return r;
}

// B matrix 32x16 bf16 fed as B^T row-major ([n][k]): col = lane&15;
// lanes 0-15 hold K 0..7 & 16..23, lanes 16-31 hold K 8..15 & 24..31.  Per-lane pointer:
//   p = tile_origin + (lane&15)*ld + ((lane>>4)<<3)
DEVFN v16bf frag_bt(const __bf16* p) {
  v8bf lo = *(const v8bf*)(p);        // K = kb .. kb+7
  v8bf hi = *(const v8bf*)(p + 16);   // K = kb+16 .. kb+23
  v16bf r;
#pragma unroll
  for (int i = 0; i < 8; ++i) { r[i] = lo[i]; r[i + 8] = hi[i]; }
  return r;
}

DEVFN v8f wmma_bf16(v16bf a, v16bf b, v8f c) {
  // 8 args: (neg_a, A, neg_b, B, c_mod, C, reuse_a, reuse_b)
  return __builtin_amdgcn_wmma_f32_16x16x32_bf16(
      false, a, false, b, (short)0, c, false, false);
}

// C/D f32 16x16 layout: col n = lane&15, rows m = j + 8*(lane>>4), j=0..7.
// MODE 0: bf16 row-major, MODE 1: bf16 transposed (out[n*ldc+m]), MODE 2: f32.
template <int MODE>
DEVFN void store_acc(v8f acc, int m0, int n0, void* out, int ldc,
                     float scale, const float* bias) {
  const int lane = threadIdx.x & 31;
  const int n  = n0 + (lane & 15);
  const int mb = m0 + ((lane >> 4) << 3);
  const float bv = bias ? bias[n] : 0.0f;
  if (MODE == 2) {
    float* o = (float*)out;
#pragma unroll
    for (int j = 0; j < 8; ++j)
      o[(size_t)(mb + j) * ldc + n] = acc[j] * scale + bv;
  } else if (MODE == 0) {
    __bf16* o = (__bf16*)out;
#pragma unroll
    for (int j = 0; j < 8; ++j)
      o[(size_t)(mb + j) * ldc + n] = (__bf16)(acc[j] * scale + bv);
  } else {  // transposed bf16 (used to build V^T); contiguous in j per lane
    __bf16* o = (__bf16*)out;
#pragma unroll
    for (int j = 0; j < 8; ++j)
      o[(size_t)n * ldc + (mb + j)] = (__bf16)(acc[j] * scale + bv);
  }
}

// --- Generic bf16 GEMM: C = scale * (A @ B) + bias --------------------------
// A: [M,K] row-major bf16 (lda). Bt: B^T row-major [N,K] bf16 (ldb).
// One wave per 32x64 output block (M%32==0, N%64==0 for every GEMM here).
template <int MODE>
__global__ __launch_bounds__(256) void gemm_bf16(
    const __bf16* __restrict__ A, int lda,
    const __bf16* __restrict__ Bt, int ldb,
    void* __restrict__ Cout, int ldc,
    int M, int N, int K, float scale,
    const float* __restrict__ bias) {
  const int w  = (int)((blockIdx.x * blockDim.x + threadIdx.x) >> 5);
  const int tn = N >> 6;                     // 64-wide column tiles
  if (w >= (M >> 5) * tn) return;            // wave-uniform: EXEC stays full
  const int m0 = (w / tn) << 5;
  const int n0 = (w % tn) << 6;
  const int lane = threadIdx.x & 31;
  const int la = (lane & 15);
  const int ka = ((lane >> 4) << 3);         // A-frag K sub-offset (0/8)
  const int kb = ((lane >> 4) << 3);         // B-frag K sub-offset (0/8)

  // Per-lane resolved pointers; advanced by 32 elements per k-step so the
  // compiler can fold k-offsets into load immediates.
  const __bf16* pa0 = A  + (size_t)(m0 + la) * lda + ka;
  const __bf16* pa1 = pa0 + (size_t)16 * lda;
  const __bf16* pb0 = Bt + (size_t)(n0 + la) * ldb + kb;
  const __bf16* pb1 = pb0 + (size_t)16 * ldb;
  const __bf16* pb2 = pb0 + (size_t)32 * ldb;
  const __bf16* pb3 = pb0 + (size_t)48 * ldb;

  v8f acc[2][4];
#pragma unroll
  for (int i = 0; i < 2; ++i)
#pragma unroll
    for (int j = 0; j < 4; ++j) acc[i][j] = {};

  for (int k0 = 0; k0 < K; k0 += 32) {
    if (k0 + 32 < K) {  // near-scope prefetch of the next k-slab
      __builtin_prefetch(pa0 + 32, 0, 3);
      __builtin_prefetch(pa1 + 32, 0, 3);
      __builtin_prefetch(pb0 + 32, 0, 3);
      __builtin_prefetch(pb1 + 32, 0, 3);
      __builtin_prefetch(pb2 + 32, 0, 3);
      __builtin_prefetch(pb3 + 32, 0, 3);
    }
    v16bf a0 = frag_a(pa0);
    v16bf a1 = frag_a(pa1);
    v16bf b0 = frag_bt(pb0);
    v16bf b1 = frag_bt(pb1);
    v16bf b2 = frag_bt(pb2);
    v16bf b3 = frag_bt(pb3);
    acc[0][0] = wmma_bf16(a0, b0, acc[0][0]);
    acc[0][1] = wmma_bf16(a0, b1, acc[0][1]);
    acc[0][2] = wmma_bf16(a0, b2, acc[0][2]);
    acc[0][3] = wmma_bf16(a0, b3, acc[0][3]);
    acc[1][0] = wmma_bf16(a1, b0, acc[1][0]);
    acc[1][1] = wmma_bf16(a1, b1, acc[1][1]);
    acc[1][2] = wmma_bf16(a1, b2, acc[1][2]);
    acc[1][3] = wmma_bf16(a1, b3, acc[1][3]);
    pa0 += 32; pa1 += 32;
    pb0 += 32; pb1 += 32; pb2 += 32; pb3 += 32;
  }
#pragma unroll
  for (int i = 0; i < 2; ++i)
#pragma unroll
    for (int j = 0; j < 4; ++j)
      store_acc<MODE>(acc[i][j], m0 + 16 * i, n0 + 16 * j, Cout, ldc, scale,
                      bias);
}

// --- f32 -> bf16 conversion -------------------------------------------------
__global__ void cvt_f32_bf16(const float* __restrict__ s,
                             __bf16* __restrict__ d, int n) {
  int i = (int)(blockIdx.x * blockDim.x + threadIdx.x);
  if (i < n) d[i] = (__bf16)s[i];
}

// --- Column-wise softmax (reference quirk: softmax over QUERY axis) ---------
// scores: [S,S] f32 row-major. Thread t normalizes column t over all rows s
// (warp reads a contiguous 128B row segment per step -> coalesced).
__global__ __launch_bounds__(256) void softmax_col(
    const float* __restrict__ sc, __bf16* __restrict__ attn) {
  const int t = (int)(blockIdx.x * blockDim.x + threadIdx.x);
  if (t >= gS) return;
  float m = -3.0e38f, sum = 0.0f;
  for (int s = 0; s < gS; ++s) {           // online max + rescaled sum
    float v = sc[(size_t)s * gS + t];
    if (v > m) { sum = sum * __expf(m - v) + 1.0f; m = v; }
    else       { sum += __expf(v - m); }
  }
  const float inv = 1.0f / sum;
  for (int s = 0; s < gS; ++s) {
    float v = sc[(size_t)s * gS + t];
    attn[(size_t)s * gS + t] = (__bf16)(__expf(v - m) * inv);
  }
}

// ---------------------------------------------------------------------------
extern "C" void kernel_launch(void* const* d_in, const int* in_sizes, int n_in,
                              void* d_out, int out_size, void* d_ws, size_t ws_size,
                              hipStream_t stream) {
  (void)in_sizes; (void)n_in; (void)out_size; (void)ws_size;
  const float* x  = (const float*)d_in[0];
  const float* Wq = (const float*)d_in[1];
  const float* bq = (const float*)d_in[2];
  const float* Wk = (const float*)d_in[3];
  const float* bk = (const float*)d_in[4];
  const float* Wv = (const float*)d_in[5];
  const float* bv = (const float*)d_in[6];
  const float* Wo = (const float*)d_in[7];
  const float* bo = (const float*)d_in[8];
  float* out = (float*)d_out;

  // Workspace carve-up (~67 MB total; per-(b,h) buffers are reused across the
  // serialized launch loop to keep scratch small).
  char* ws = (char*)d_ws;
  size_t off = 0;
  auto carve = [&](size_t bytes) -> char* {
    char* p = ws + off;
    off = (off + bytes + 255) & ~(size_t)255;
    return p;
  };
  __bf16* xbf  = (__bf16*)carve((size_t)gB * gS * gD * 2);   // [B*S, D]
  __bf16* wqb  = (__bf16*)carve((size_t)gH * gD * gD * 2);   // [h][e][d]
  __bf16* wkb  = (__bf16*)carve((size_t)gH * gD * gD * 2);
  __bf16* wvb  = (__bf16*)carve((size_t)gH * gD * gD * 2);
  __bf16* wob  = (__bf16*)carve((size_t)gD * gHD * 2);       // [o][f]
  __bf16* qbf  = (__bf16*)carve((size_t)gS * gD * 2);        // per (b,h)
  __bf16* kbf  = (__bf16*)carve((size_t)gS * gD * 2);
  __bf16* vtb  = (__bf16*)carve((size_t)gD * gS * 2);        // V^T: [e][t]
  float*  scb  = (float*) carve((size_t)gS * gS * 4);        // scores f32
  __bf16* atb  = (__bf16*)carve((size_t)gS * gS * 2);        // attn bf16
  __bf16* zbf  = (__bf16*)carve((size_t)gB * gS * gHD * 2);  // [B*S, H*D]

  auto cvt = [&](const float* s, __bf16* d, int n) {
    cvt_f32_bf16<<<(n + 255) / 256, 256, 0, stream>>>(s, d, n);
  };
  cvt(x,  xbf, gB * gS * gD);
  cvt(Wq, wqb, gH * gD * gD);
  cvt(Wk, wkb, gH * gD * gD);
  cvt(Wv, wvb, gH * gD * gD);
  cvt(Wo, wob, gD * gHD);

  // One wave per 32x64 output block, 8 waves (256 threads) per workgroup.
  const int blk_proj = ((gS / 32) * (gD / 64)) / 8;        // 32
  const int blk_sc   = ((gS / 32) * (gS / 64)) / 8;        // 256
  const int blk_out  = ((gB * gS / 32) * (gD / 64)) / 8;   // 128
  const float inv_sqrt_d = 0.0625f;  // 1/sqrt(256)

  for (int b = 0; b < gB; ++b) {
    const __bf16* xb = xbf + (size_t)b * gS * gD;
    for (int h = 0; h < gH; ++h) {
      const size_t wofs = (size_t)h * gD * gD;
      // Q = x @ Wq[h]^T + bq[h]   (bf16 out, row-major [S,D])
      gemm_bf16<0><<<blk_proj, 256, 0, stream>>>(
          xb, gD, wqb + wofs, gD, qbf, gD, gS, gD, gD, 1.0f, bq + h * gD);
      // K = x @ Wk[h]^T + bk[h]
      gemm_bf16<0><<<blk_proj, 256, 0, stream>>>(
          xb, gD, wkb + wofs, gD, kbf, gD, gS, gD, gD, 1.0f, bk + h * gD);
      // V^T = (x @ Wv[h]^T + bv[h])^T  (stored [e][t] so attn@V has fast B loads)
      gemm_bf16<1><<<blk_proj, 256, 0, stream>>>(
          xb, gD, wvb + wofs, gD, vtb, gS, gS, gD, gD, 1.0f, bv + h * gD);
      // scores = (Q @ K^T) / sqrt(D)   (f32, [S,S])
      gemm_bf16<2><<<blk_sc, 256, 0, stream>>>(
          qbf, gD, kbf, gD, scb, gS, gS, gS, gD, inv_sqrt_d, nullptr);
      // attn = softmax over query axis (column-wise), bf16
      softmax_col<<<gS / 256, 256, 0, stream>>>(scb, atb);
      // Z[:, h*D:(h+1)*D] = attn @ V  (bf16 into head-major concat buffer)
      gemm_bf16<0><<<blk_proj, 256, 0, stream>>>(
          atb, gS, vtb, gS, zbf + (size_t)b * gS * gHD + (size_t)h * gD, gHD,
          gS, gD, gS, 1.0f, nullptr);
    }
  }
  // out = Z @ Wo^T + bo   (f32, [B*S, D])
  gemm_bf16<2><<<blk_out, 256, 0, stream>>>(
      zbf, gHD, wob, gHD, out, gD, gB * gS, gD, gHD, 1.0f, bo);
}
